// RoIHeads_48163763258067
// MI455X (gfx1250) — matrix operand order repaired and checked by reading, not compiled, not run on hardware
//
#include <hip/hip_runtime.h>
#include <hip/hip_bf16.h>
#include <hip/hip_fp16.h>

// ---------- problem constants ----------
#define Bn 4
#define Np 256
#define Cc 256
#define FH 100
#define FW 152
#define IMG_H 800.0f
#define IMG_W 1216.0f
#define SCALE 0.125f
#define POOL 7
#define DIN 12544          // C*POOL*POOL
#define H1 1024
#define NCLS 91
#define NHEAD 512          // padded 91 + 364 -> 512
#define NCAND 23040        // Np * 90
#define PRE_NMS 1024
#define DETS 100
#define SCORE_T 0.05f
#define NMS_T 0.5f
#define MIN_SIZE 0.01f
#define BBOX_CLIP 4.135166556742356f
#define WXY 10.0f
#define WWH 5.0f

typedef __attribute__((ext_vector_type(16))) _Float16 v16h;
typedef __attribute__((ext_vector_type(8)))  float    v8f;
typedef int v4i __attribute__((vector_size(16)));
typedef v4i __attribute__((address_space(1)))* v4i_gptr;   // global
typedef v4i __attribute__((address_space(3)))* v4i_lptr;   // LDS

// gfx1250 async global->LDS path (guarded: falls back to uint4 staging)
#if defined(__has_builtin)
#if __has_builtin(__builtin_amdgcn_global_load_async_to_lds_b128) && \
    __has_builtin(__builtin_amdgcn_s_wait_asynccnt)
#define HAVE_ASYNC_LDS 1
#endif
#endif
#ifndef HAVE_ASYNC_LDS
#define HAVE_ASYNC_LDS 0
#endif

// ---------- weight conversion ----------
__global__ void f32_to_f16_kernel(const float* __restrict__ src,
                                  _Float16* __restrict__ dst, int n) {
  for (int i = blockIdx.x * blockDim.x + threadIdx.x; i < n;
       i += gridDim.x * blockDim.x)
    dst[i] = (_Float16)src[i];
}

__global__ void pack_heads_kernel(const float* __restrict__ Wc,
                                  const float* __restrict__ bc,
                                  const float* __restrict__ Wb,
                                  const float* __restrict__ bb,
                                  _Float16* __restrict__ WH,
                                  float* __restrict__ bH) {
  int total = H1 * NHEAD;
  for (int i = blockIdx.x * blockDim.x + threadIdx.x; i < total;
       i += gridDim.x * blockDim.x) {
    int k = i / NHEAD, j = i % NHEAD;
    float v = 0.0f;
    if (j < NCLS)            v = Wc[k * NCLS + j];
    else if (j < NCLS + 364) v = Wb[k * 364 + (j - NCLS)];
    WH[i] = (_Float16)v;
    if (i < NHEAD) {
      float bv = 0.0f;
      if (i < NCLS)            bv = bc[i];
      else if (i < NCLS + 364) bv = bb[i - NCLS];
      bH[i] = bv;
    }
  }
}

// ---------- RoI Align (block = one RoI) ----------
__global__ __launch_bounds__(256) void roi_align_kernel(
    const float* __restrict__ feat, const float* __restrict__ rois,
    _Float16* __restrict__ Xh) {
  int r = blockIdx.x;            // 0..1023
  int b = r / Np;
  int tid = threadIdx.x;

  __shared__ int   sx0[14], sx1[14], sy0[14], sy1[14];
  __shared__ float slx[14], sly[14];

  const float* roi = rois + (size_t)r * 4;
  float x1 = roi[0] * SCALE, y1 = roi[1] * SCALE;
  float x2 = roi[2] * SCALE, y2 = roi[3] * SCALE;
  float rw = fmaxf(x2 - x1, 1.0f), rh = fmaxf(y2 - y1, 1.0f);

  if (tid < 28) {
    int s = tid % 14;
    bool isY = tid >= 14;
    float g  = (float)(s >> 1) + 0.25f + 0.5f * (float)(s & 1);
    float lo = isY ? y1 : x1;
    float rr = isY ? rh : rw;
    float mx = isY ? (float)(FH - 1) : (float)(FW - 1);
    float gc = fminf(fmaxf(lo + (rr / (float)POOL) * g, 0.0f), mx);
    float f0 = floorf(gc);
    float l  = gc - f0;
    int i0 = (int)fminf(fmaxf(f0, 0.0f), mx);
    int i1 = (int)fminf(fmaxf(f0 + 1.0f, 0.0f), mx);
    if (isY) { sy0[s] = i0; sy1[s] = i1; sly[s] = l; }
    else     { sx0[s] = i0; sx1[s] = i1; slx[s] = l; }
  }
  __syncthreads();

  const float* fb_ = feat + (size_t)b * Cc * FH * FW;
  for (int idx = tid; idx < Cc * 49; idx += 256) {
    int c = idx / 49, cell = idx % 49;
    int py = cell / 7, px = cell % 7;
    const float* fc = fb_ + (size_t)c * FH * FW;
    float acc = 0.0f;
#pragma unroll
    for (int sy = 0; sy < 2; ++sy) {
      int iy = py * 2 + sy;
      int y0 = sy0[iy], y1i = sy1[iy];
      float ly = sly[iy];
#pragma unroll
      for (int sx = 0; sx < 2; ++sx) {
        int ix = px * 2 + sx;
        int x0 = sx0[ix], x1i = sx1[ix];
        float lx = slx[ix];
        float v00 = fc[y0 * FW + x0],  v01 = fc[y0 * FW + x1i];
        float v10 = fc[y1i * FW + x0], v11 = fc[y1i * FW + x1i];
        acc += (1.0f - ly) * (1.0f - lx) * v00 + (1.0f - ly) * lx * v01 +
               ly * (1.0f - lx) * v10 + ly * lx * v11;
      }
    }
    Xh[(size_t)r * DIN + idx] = (_Float16)(acc * 0.25f);
  }
}

// ---------- WMMA GEMM: out = [relu](A[MxK] * B[KxN] + bias) ----------
// Block tile 128x128, 8 waves, each wave 32x64 = 2x4 WMMA accumulators.
// A staged row-major (async-to-LDS when available); B staged TRANSPOSED so
// both A and B fragments are K-contiguous per lane -> paired 32b ds loads.
#define BM 128
#define BN 128
#define KB 32
#define APAD 8   // A row = 40 halves = 80B -> every 16B segment aligned
#define BPAD 2   // BsT row = 34 halves = 68B -> 17-dword stride, bank-clean
__global__ __launch_bounds__(256) void gemm_wmma_kernel(
    const _Float16* __restrict__ A, const _Float16* __restrict__ Bm,
    const float* __restrict__ bias, int M, int N, int K,
    _Float16* __restrict__ outH, float* __restrict__ outF, int applyRelu) {
  __shared__ _Float16 As[BM][KB + APAD];
  __shared__ _Float16 BsT[BN][KB + BPAD];   // [col][k]

  int tid  = threadIdx.x;
  int wave = tid >> 5;
  int lane = tid & 31;
  int wr = wave >> 1;           // 0..3 -> 32-row strip
  int wc = wave & 1;            // 0..1 -> 64-col strip
  int m0 = blockIdx.y * BM;
  int n0 = blockIdx.x * BN;

  v8f acc[2][4] = {};

  int lhalf = (lane >= 16) ? 1 : 0;
  int lrow  = lane & 15;

  for (int k0 = 0; k0 < K; k0 += KB) {
    // prefetch next K-chunk (gfx1250 global_prefetch_b8 path)
    if (k0 + KB < K) {
      __builtin_prefetch(A + (size_t)(m0 + (tid >> 2)) * K + k0 + KB, 0, 3);
      __builtin_prefetch(Bm + (size_t)(k0 + KB + (tid >> 4)) * N + n0, 0, 3);
    }

    // ---- stage A tile: 128 rows x 32 halves = 512 x 16B segments ----
#pragma unroll
    for (int j = 0; j < 2; ++j) {
      int s   = tid + 256 * j;
      int row = s >> 2;
      int seg = (s & 3) * 8;                      // half offset in row
      const _Float16* gp = A + (size_t)(m0 + row) * K + k0 + seg;
      _Float16* lp = &As[row][seg];
#if HAVE_ASYNC_LDS
      __builtin_amdgcn_global_load_async_to_lds_b128(
          (v4i_gptr)(uintptr_t)gp, (v4i_lptr)lp, 0, 0);
#else
      *reinterpret_cast<uint4*>(lp) =
          *reinterpret_cast<const uint4*>(gp);
#endif
    }

    // ---- stage B tile transposed: 32 rows x 128 halves ----
#pragma unroll
    for (int j = 0; j < 2; ++j) {
      int s    = tid + 256 * j;
      int row  = s >> 4;                          // k row 0..31
      int coff = (s & 15) * 8;                    // col offset 0..120
      union { uint4 v; _Float16 h[8]; } u;
      u.v = *reinterpret_cast<const uint4*>(Bm + (size_t)(k0 + row) * N +
                                            n0 + coff);
#pragma unroll
      for (int i = 0; i < 8; ++i) BsT[coff + i][row] = u.h[i];
    }

#if HAVE_ASYNC_LDS
    __builtin_amdgcn_s_wait_asynccnt(0);
#endif
    __syncthreads();

    // ---- fragments (wave32): K-contiguous per lane, k = 16*half + i for both A and B ----
    v16h a0, a1, bfr[4];
#pragma unroll
    for (int i = 0; i < 16; ++i) {
      int ka = lhalf * 16 + i;                              // A K index
      a0[i] = As[wr * 32 + lrow][ka];
      a1[i] = As[wr * 32 + 16 + lrow][ka];
      int kb = lhalf * 16 + i;                              // B K index
#pragma unroll
      for (int t = 0; t < 4; ++t)
        bfr[t][i] = BsT[wc * 64 + t * 16 + lrow][kb];
    }
#pragma unroll
    for (int t = 0; t < 4; ++t) {
      acc[0][t] = __builtin_amdgcn_wmma_f32_16x16x32_f16(
          false, a0, false, bfr[t], (short)0, acc[0][t], false, false);
      acc[1][t] = __builtin_amdgcn_wmma_f32_16x16x32_f16(
          false, a1, false, bfr[t], (short)0, acc[1][t], false, false);
    }
    __syncthreads();
  }

  // epilogue: C layout — vgpr v: M = v (+8 for lanes>=16); lane&15 = N
  int lhalf8 = lhalf * 8;
#pragma unroll
  for (int sa = 0; sa < 2; ++sa) {
#pragma unroll
    for (int sb = 0; sb < 4; ++sb) {
      int col  = n0 + wc * 64 + sb * 16 + lrow;
      int row0 = m0 + wr * 32 + sa * 16 + lhalf8;
      float bv = bias ? bias[col] : 0.0f;
#pragma unroll
      for (int v = 0; v < 8; ++v) {
        float x = acc[sa][sb][v] + bv;
        if (applyRelu) x = fmaxf(x, 0.0f);
        size_t o = (size_t)(row0 + v) * N + col;
        if (outH) outH[o] = (_Float16)x;
        if (outF) outF[o] = x;
      }
    }
  }
}

// ---------- decode + softmax (block = one RoI) ----------
__global__ __launch_bounds__(128) void decode_softmax_kernel(
    const float* __restrict__ S, const float* __restrict__ rois,
    float* __restrict__ candBoxes, float* __restrict__ candScores) {
  int r = blockIdx.x;          // 0..1023
  int b = r / Np, n = r % Np;
  int t = threadIdx.x;
  __shared__ float red[128];

  float logit = (t < NCLS) ? S[(size_t)r * NHEAD + t] : -1e30f;
  red[t] = logit;
  __syncthreads();
  for (int s = 64; s > 0; s >>= 1) {
    if (t < s) red[t] = fmaxf(red[t], red[t + s]);
    __syncthreads();
  }
  float mx = red[0];
  __syncthreads();
  float e = (t < NCLS) ? expf(logit - mx) : 0.0f;
  red[t] = e;
  __syncthreads();
  for (int s = 64; s > 0; s >>= 1) {
    if (t < s) red[t] += red[t + s];
    __syncthreads();
  }
  float inv = 1.0f / red[0];

  if (t >= 1 && t <= 90) {
    int j = t;                                   // class index 1..90
    float score = e * inv;
    const float* roi = rois + (size_t)r * 4;
    float wR = roi[2] - roi[0], hR = roi[3] - roi[1];
    float cx = roi[0] + 0.5f * wR, cy = roi[1] + 0.5f * hR;
    const float* d = S + (size_t)r * NHEAD + NCLS + j * 4;
    float dx = d[0] / WXY, dy = d[1] / WXY;
    float dw = fminf(d[2] / WWH, BBOX_CLIP);
    float dh = fminf(d[3] / WWH, BBOX_CLIP);
    float pcx = dx * wR + cx, pcy = dy * hR + cy;
    float pw = expf(dw) * wR, ph = expf(dh) * hR;
    float bx1 = fminf(fmaxf(pcx - 0.5f * pw, 0.0f), IMG_W);
    float by1 = fminf(fmaxf(pcy - 0.5f * ph, 0.0f), IMG_H);
    float bx2 = fminf(fmaxf(pcx + 0.5f * pw, 0.0f), IMG_W);
    float by2 = fminf(fmaxf(pcy + 0.5f * ph, 0.0f), IMG_H);
    float ws = bx2 - bx1, hs = by2 - by1;
    bool valid = (score > SCORE_T) && (ws >= MIN_SIZE) && (hs >= MIN_SIZE);
    int ci = n * 90 + (j - 1);
    candScores[(size_t)b * NCAND + ci] = valid ? score : -1.0f;
    float* cb = candBoxes + ((size_t)b * NCAND + ci) * 4;
    cb[0] = bx1; cb[1] = by1; cb[2] = bx2; cb[3] = by2;
  }
}

// ---------- top-k + NMS + top-100 (block = one image) ----------
#define CHUNK 4096
__global__ __launch_bounds__(1024) void nms_kernel(
    const float* __restrict__ candBoxes, const float* __restrict__ candScores,
    float* __restrict__ outBoxes, float* __restrict__ outScores,
    float* __restrict__ outLabels) {
  int b = blockIdx.x;
  int tid = threadIdx.x;

  __shared__ float chunk[CHUNK];
  __shared__ int   selIdx[PRE_NMS];
  __shared__ float selVal[PRE_NMS];
  __shared__ float obx1[PRE_NMS], oby1[PRE_NMS], obx2[PRE_NMS], oby2[PRE_NMS];
  __shared__ int   keepS[PRE_NMS];

  const float* fs = candScores + (size_t)b * NCAND;

  selVal[tid] = -1.0f;
  selIdx[tid] = 0;

  // exact stable top-1024 via rank counting (ties broken by lower index)
  float myV[23];
  int   myRank[23];
  int nMine = 0;
  for (int i = tid; i < NCAND; i += 1024) {
    myV[nMine] = fs[i];
    myRank[nMine] = 0;
    nMine++;
  }
  for (int c0 = 0; c0 < NCAND; c0 += CHUNK) {
    __syncthreads();
    int lim = min(CHUNK, NCAND - c0);
    for (int j = tid; j < lim; j += 1024) chunk[j] = fs[c0 + j];
    __syncthreads();
    for (int m = 0; m < nMine; ++m) {
      float v = myV[m];
      int gi = tid + m * 1024;
      int rk = 0;
      for (int j = 0; j < lim; ++j) {
        float o = chunk[j];
        int jg = c0 + j;
        rk += (o > v) || (o == v && jg < gi);
      }
      myRank[m] += rk;
    }
  }
  __syncthreads();
  for (int m = 0; m < nMine; ++m)
    if (myRank[m] < PRE_NMS) {
      selIdx[myRank[m]] = tid + m * 1024;
      selVal[myRank[m]] = myV[m];
    }
  __syncthreads();

  // label-offset boxes + initial keep
  {
    int i = selIdx[tid];
    const float* cb = candBoxes + ((size_t)b * NCAND + i) * 4;
    float lab = (float)((i % 90) + 1);
    float off = lab * (IMG_W + 2.0f);      // max(IMG_H,IMG_W)+2
    obx1[tid] = cb[0] + off; oby1[tid] = cb[1] + off;
    obx2[tid] = cb[2] + off; oby2[tid] = cb[3] + off;
    keepS[tid] = (selVal[tid] > SCORE_T) ? 1 : 0;
  }
  __syncthreads();

  float ax1 = obx1[tid], ay1 = oby1[tid], ax2 = obx2[tid], ay2 = oby2[tid];
  float areaA = (ax2 - ax1) * (ay2 - ay1);
  for (int i = 0; i < PRE_NMS; ++i) {
    if (keepS[i] && tid > i) {
      float bx1 = obx1[i], by1 = oby1[i], bx2 = obx2[i], by2 = oby2[i];
      float areaB = (bx2 - bx1) * (by2 - by1);
      float ltx = fmaxf(ax1, bx1), lty = fmaxf(ay1, by1);
      float rbx = fminf(ax2, bx2), rby = fminf(ay2, by2);
      float iw = fmaxf(rbx - ltx, 0.0f), ih = fmaxf(rby - lty, 0.0f);
      float inter = iw * ih;
      float iou = inter / (areaA + areaB - inter + 1e-9f);
      if (iou > NMS_T) keepS[tid] = 0;
    }
    __syncthreads();
  }

  // zero-fill outputs
  if (tid < DETS) {
    size_t s = (size_t)b * DETS + tid;
    outBoxes[s * 4 + 0] = 0.0f; outBoxes[s * 4 + 1] = 0.0f;
    outBoxes[s * 4 + 2] = 0.0f; outBoxes[s * 4 + 3] = 0.0f;
    outScores[s] = 0.0f;
    outLabels[s] = 0.0f;
  }
  __syncthreads();

  // compact first 100 kept (selVal is descending, so this is top-100)
  int kc = 0;
  for (int j = 0; j < tid; ++j) kc += keepS[j];
  if (keepS[tid] && kc < DETS) {
    size_t s = (size_t)b * DETS + kc;
    int i = selIdx[tid];
    const float* cb = candBoxes + ((size_t)b * NCAND + i) * 4;
    outBoxes[s * 4 + 0] = cb[0]; outBoxes[s * 4 + 1] = cb[1];
    outBoxes[s * 4 + 2] = cb[2]; outBoxes[s * 4 + 3] = cb[3];
    outScores[s] = selVal[tid];
    outLabels[s] = (float)((i % 90) + 1);
  }
}

// ---------- host ----------
extern "C" void kernel_launch(void* const* d_in, const int* in_sizes, int n_in,
                              void* d_out, int out_size, void* d_ws,
                              size_t ws_size, hipStream_t stream) {
  const float* features  = (const float*)d_in[0];
  const float* proposals = (const float*)d_in[1];
  const float* W6 = (const float*)d_in[2];
  const float* b6 = (const float*)d_in[3];
  const float* W7 = (const float*)d_in[4];
  const float* b7 = (const float*)d_in[5];
  const float* Wc = (const float*)d_in[6];
  const float* bc = (const float*)d_in[7];
  const float* Wb = (const float*)d_in[8];
  const float* bb = (const float*)d_in[9];
  float* out = (float*)d_out;

  char* p = (char*)d_ws;
  auto alloc = [&](size_t bytes) {
    void* r = (void*)p;
    p += (bytes + 255) & ~(size_t)255;
    return r;
  };
  const int R = Bn * Np;                                    // 1024 RoIs
  _Float16* W6h = (_Float16*)alloc((size_t)DIN * H1 * 2);
  _Float16* W7h = (_Float16*)alloc((size_t)H1 * H1 * 2);
  _Float16* WHh = (_Float16*)alloc((size_t)H1 * NHEAD * 2);
  float*    bH  = (float*)alloc((size_t)NHEAD * 4);
  _Float16* Xh  = (_Float16*)alloc((size_t)R * DIN * 2);
  _Float16* X1h = (_Float16*)alloc((size_t)R * H1 * 2);
  _Float16* X2h = (_Float16*)alloc((size_t)R * H1 * 2);
  float*    S   = (float*)alloc((size_t)R * NHEAD * 4);
  float* candBoxes  = (float*)alloc((size_t)Bn * NCAND * 4 * 4);
  float* candScores = (float*)alloc((size_t)Bn * NCAND * 4);
  (void)ws_size; (void)n_in; (void)in_sizes; (void)out_size;

  // 1) weight conversion / packing
  f32_to_f16_kernel<<<2048, 256, 0, stream>>>(W6, W6h, DIN * H1);
  f32_to_f16_kernel<<<1024, 256, 0, stream>>>(W7, W7h, H1 * H1);
  pack_heads_kernel<<<512, 256, 0, stream>>>(Wc, bc, Wb, bb, WHh, bH);

  // 2) RoI-Align
  roi_align_kernel<<<R, 256, 0, stream>>>(features, proposals, Xh);

  // 3) FC6 -> FC7 -> heads (WMMA GEMMs)
  gemm_wmma_kernel<<<dim3(H1 / BN, R / BM), 256, 0, stream>>>(
      Xh, W6h, b6, R, H1, DIN, X1h, nullptr, 1);
  gemm_wmma_kernel<<<dim3(H1 / BN, R / BM), 256, 0, stream>>>(
      X1h, W7h, b7, R, H1, H1, X2h, nullptr, 1);
  gemm_wmma_kernel<<<dim3(NHEAD / BN, R / BM), 256, 0, stream>>>(
      X2h, WHh, bH, R, NHEAD, H1, nullptr, S, 0);

  // 4) decode + softmax
  decode_softmax_kernel<<<R, 128, 0, stream>>>(S, proposals, candBoxes,
                                               candScores);

  // 5) per-image top-k + NMS + top-100
  float* outBoxes  = out;                 // B*100*4
  float* outScores = out + Bn * DETS * 4; // B*100
  float* outLabels = out + Bn * DETS * 5; // B*100
  nms_kernel<<<Bn, 1024, 0, stream>>>(candBoxes, candScores, outBoxes,
                                      outScores, outLabels);
}
